// MultiheadAttention_65395172049178
// MI455X (gfx1250) — matrix operand order repaired and checked by reading, not compiled, not run on hardware
//
#include <hip/hip_runtime.h>

// ---------------------------------------------------------------------------
// MultiheadAttention for MI455X (gfx1250, wave32, WMMA 16x16x32 bf16)
// B=4, H=16, M=N=2048, E=1024, D=64, SCALE=1/8
// ---------------------------------------------------------------------------

typedef __attribute__((ext_vector_type(16))) __bf16 v16bf;
typedef __attribute__((ext_vector_type(8)))  __bf16 v8bf;
typedef __attribute__((ext_vector_type(8)))  float  v8f;

union BF16Frag {
    v16bf v;
    v8bf  h[2];
};

constexpr int Bc = 4;     // batch
constexpr int Hh = 16;    // heads
constexpr int Mm = 2048;  // query len
constexpr int Nn = 2048;  // key len
constexpr int Ee = 1024;  // embed
constexpr int Dd = 64;    // head dim
constexpr float SCALE = 0.125f;  // 1/sqrt(64)

// fp32 -> bf16 round-to-nearest-even
static __device__ __forceinline__ unsigned short f2bf(float f) {
    union { float f; unsigned u; } x;
    x.f = f;
    unsigned r = x.u + 0x7FFFu + ((x.u >> 16) & 1u);
    return (unsigned short)(r >> 16);
}
// pack two floats into one dword of bf16 {lo, hi} (matches WMMA K-pair order)
static __device__ __forceinline__ unsigned pack2(float lo, float hi) {
    return (unsigned)f2bf(lo) | ((unsigned)f2bf(hi) << 16);
}

// ---- 16-lane butterfly reductions via DPP16 (pure VALU, no LDS pipe) ------
static __device__ __forceinline__ float dpp_max16(float x) {
    int xi;
    xi = __builtin_amdgcn_update_dpp(0, __builtin_bit_cast(int, x), 0xB1, 0xF, 0xF, true);  // quad_perm(1,0,3,2)
    x  = fmaxf(x, __builtin_bit_cast(float, xi));
    xi = __builtin_amdgcn_update_dpp(0, __builtin_bit_cast(int, x), 0x4E, 0xF, 0xF, true);  // quad_perm(2,3,0,1)
    x  = fmaxf(x, __builtin_bit_cast(float, xi));
    xi = __builtin_amdgcn_update_dpp(0, __builtin_bit_cast(int, x), 0x141, 0xF, 0xF, true); // row_half_mirror
    x  = fmaxf(x, __builtin_bit_cast(float, xi));
    xi = __builtin_amdgcn_update_dpp(0, __builtin_bit_cast(int, x), 0x140, 0xF, 0xF, true); // row_mirror
    x  = fmaxf(x, __builtin_bit_cast(float, xi));
    return x;
}
static __device__ __forceinline__ float dpp_add16(float x) {
    int xi;
    xi = __builtin_amdgcn_update_dpp(0, __builtin_bit_cast(int, x), 0xB1, 0xF, 0xF, true);
    x += __builtin_bit_cast(float, xi);
    xi = __builtin_amdgcn_update_dpp(0, __builtin_bit_cast(int, x), 0x4E, 0xF, 0xF, true);
    x += __builtin_bit_cast(float, xi);
    xi = __builtin_amdgcn_update_dpp(0, __builtin_bit_cast(int, x), 0x141, 0xF, 0xF, true);
    x += __builtin_bit_cast(float, xi);
    xi = __builtin_amdgcn_update_dpp(0, __builtin_bit_cast(int, x), 0x140, 0xF, 0xF, true);
    x += __builtin_bit_cast(float, xi);
    return x;
}

// ---------------------------------------------------------------------------
// Linear: Out = alpha * (A @ W + bias)
//   MODE 0: A fp32 at r*E (r = m*B+b), out bf16 scattered [b,h,seq,d]  (Q,K)
//   MODE 2: A fp32 at r*E,             out bf16 scattered [b,h,d,seq]  (V, pre-transposed)
//   MODE 1: A bf16 at [b][m][E],       out fp32 at r*E (d_out (M,B,E))
// Block: 256 threads (8 waves), tile 128 rows x 64 cols, K-step 32.
// Register double-buffering: load tile k+1 from global while WMMAs run on k.
// ---------------------------------------------------------------------------
template <int MODE>
__device__ __forceinline__ void load_tile(const void* __restrict__ Ain,
                                          const float* __restrict__ W, int r0,
                                          int c0, int k0, int tid,
                                          float4 aF[4], uint4 aB[2],
                                          float4& b0, float4& b1) {
    int idxA = tid * 16;
    int row = idxA >> 5, kk = idxA & 31;
    if constexpr (MODE != 1) {
        const float4* src = (const float4*)((const float*)Ain +
                            (size_t)(r0 + row) * Ee + k0 + kk);
#pragma unroll
        for (int i = 0; i < 4; i++) aF[i] = src[i];
    } else {
        int gr = r0 + row;
        int m = gr >> 2, b = gr & 3;  // r = m*B+b
        const uint4* src = (const uint4*)((const unsigned short*)Ain +
                           (size_t)(b * Mm + m) * Ee + k0 + kk);
        aB[0] = src[0];
        aB[1] = src[1];
    }
    int u = tid * 4;
    int k2 = u >> 6, cb = u & 63;
    const float* w0 = W + (size_t)(k0 + 2 * k2) * Ee + c0 + cb;
    b0 = *(const float4*)w0;
    b1 = *(const float4*)(w0 + Ee);
}

template <int MODE>
__device__ __forceinline__ void store_tile(unsigned* lds_a32, unsigned* lds_b32,
                                           int tid, const float4 aF[4],
                                           const uint4 aB[2], float4 b0,
                                           float4 b1) {
    int idxA = tid * 16;
    int row = idxA >> 5, kk = idxA & 31;
    if constexpr (MODE != 1) {
        uint4 u0, u1;
        u0.x = pack2(aF[0].x, aF[0].y); u0.y = pack2(aF[0].z, aF[0].w);
        u0.z = pack2(aF[1].x, aF[1].y); u0.w = pack2(aF[1].z, aF[1].w);
        u1.x = pack2(aF[2].x, aF[2].y); u1.y = pack2(aF[2].z, aF[2].w);
        u1.z = pack2(aF[3].x, aF[3].y); u1.w = pack2(aF[3].z, aF[3].w);
        *(uint4*)&lds_a32[row * 16 + (kk >> 1)]     = u0;
        *(uint4*)&lds_a32[row * 16 + (kk >> 1) + 4] = u1;
    } else {
        *(uint4*)&lds_a32[row * 16 + (kk >> 1)]     = aB[0];
        *(uint4*)&lds_a32[row * 16 + (kk >> 1) + 4] = aB[1];
    }
    int u = tid * 4;
    int k2 = u >> 6, cb = u & 63;
    lds_b32[(cb + 0) * 16 + k2] = pack2(b0.x, b1.x);
    lds_b32[(cb + 1) * 16 + k2] = pack2(b0.y, b1.y);
    lds_b32[(cb + 2) * 16 + k2] = pack2(b0.z, b1.z);
    lds_b32[(cb + 3) * 16 + k2] = pack2(b0.w, b1.w);
}

template <int MODE>
__global__ __launch_bounds__(256) void linear_wmma(
    const void* __restrict__ Ain, const float* __restrict__ W,
    const float* __restrict__ bias, void* __restrict__ Out, float alpha) {
    __shared__ __align__(16) unsigned lds_a32[128 * 16];  // [row][k-pair]
    __shared__ __align__(16) unsigned lds_b32[64 * 16];   // [col][k-pair]

    const int tid  = threadIdx.x;
    const int wave = tid >> 5;
    const int lane = tid & 31;
    const int hi   = lane >> 4;
    const int ln   = lane & 15;
    const int c0   = blockIdx.x * 64;
    const int r0   = blockIdx.y * 128;

    const unsigned short* la = (const unsigned short*)lds_a32;
    const unsigned short* lb = (const unsigned short*)lds_b32;

    v8f acc[4];
#pragma unroll
    for (int j = 0; j < 4; j++)
#pragma unroll
        for (int i = 0; i < 8; i++) acc[j][i] = 0.0f;

    float4 aF[4];
    uint4  aB[2];
    float4 bR0, bR1;
    load_tile<MODE>(Ain, W, r0, c0, 0, tid, aF, aB, bR0, bR1);

    for (int k0 = 0; k0 < Ee; k0 += 32) {
        __syncthreads();
        store_tile<MODE>(lds_a32, lds_b32, tid, aF, aB, bR0, bR1);
        __syncthreads();
        if (k0 + 32 < Ee)  // prefetch next tile; overlaps with WMMAs below
            load_tile<MODE>(Ain, W, r0, c0, k0 + 32, tid, aF, aB, bR0, bR1);

        BF16Frag a;
        int arow = wave * 16 + ln;
        // A layout: lanes<16: K 0..7 & 16..23; lanes>=16: K 8..15 & 24..31
        a.h[0] = *(const v8bf*)&la[arow * 32 + (hi ? 8 : 0)];
        a.h[1] = *(const v8bf*)&la[arow * 32 + 16 + (hi ? 8 : 0)];
        BF16Frag bfr[4];
#pragma unroll
        for (int j = 0; j < 4; j++) {
            int bcol = j * 16 + ln;
            // B layout: lanes<16: K 0..7 & 16..23; lanes>=16: K 8..15 & 24..31
            bfr[j].h[0] = *(const v8bf*)&lb[bcol * 32 + (hi ? 8 : 0)];
            bfr[j].h[1] = *(const v8bf*)&lb[bcol * 32 + 16 + (hi ? 8 : 0)];
        }
#pragma unroll
        for (int j = 0; j < 4; j++)
            acc[j] = __builtin_amdgcn_wmma_f32_16x16x32_bf16(
                false, a.v, false, bfr[j].v, (short)0, acc[j], false, false);
    }

    // ---- epilogue: bias + alpha, store ----
#pragma unroll
    for (int j = 0; j < 4; j++) {
        int col = c0 + j * 16 + ln;
        float bv = bias[col];
#pragma unroll
        for (int i = 0; i < 8; i++) {
            int row = r0 + wave * 16 + (hi ? i + 8 : i);
            float val = (acc[j][i] + bv) * alpha;
            if constexpr (MODE == 0) {
                int m = row >> 2, b = row & 3;
                int h = col >> 6, d = col & 63;
                ((unsigned short*)Out)[((size_t)(b * Hh + h) * Mm + m) * Dd + d] =
                    f2bf(val);
            } else if constexpr (MODE == 2) {
                int m = row >> 2, b = row & 3;
                int h = col >> 6, d = col & 63;
                ((unsigned short*)Out)[((size_t)(b * Hh + h) * Dd + d) * Nn + m] =
                    f2bf(val);
            } else {
                ((float*)Out)[(size_t)row * Ee + col] = val;
            }
        }
    }
}

// ---------------------------------------------------------------------------
// Flash attention: one block per (b,h, 128-query tile); 8 independent waves,
// each owns a 16-row query strip. Q, K, and (pre-transposed) V fragments all
// load straight from global as contiguous 16B runs -- no block barriers.
// Only LDS use: per-wave C-layout -> A-layout shuffle of P, synchronized with
// a wave-local s_wait_dscnt. Softmax reductions are DPP16 butterflies.
// ---------------------------------------------------------------------------
__global__ __launch_bounds__(256) void flash_attn(
    const unsigned short* __restrict__ qb, const unsigned short* __restrict__ kb,
    const unsigned short* __restrict__ vb, unsigned short* __restrict__ ab) {
    __shared__ __align__(16) unsigned short lds_p[8 * 16 * 32];  // per-wave P

    const int tid  = threadIdx.x;
    const int wave = tid >> 5;
    const int lane = tid & 31;
    const int hi   = lane >> 4;
    const int ln   = lane & 15;
    const int mt   = blockIdx.x & 15;   // M/128 = 16 tiles
    const int bh   = blockIdx.x >> 4;   // b*H + h
    const int m0   = mt * 128;

    // ---- load Q strip fragments once (A-matrix, two K=32 steps over D=64) --
    BF16Frag aq[2];
    {
        const unsigned short* qrow =
            qb + ((size_t)bh * Mm + m0 + wave * 16 + ln) * Dd;
#pragma unroll
        for (int f = 0; f < 2; f++) {
            aq[f].h[0] = *(const v8bf*)(qrow + f * 32 + (hi ? 8 : 0));
            aq[f].h[1] = *(const v8bf*)(qrow + f * 32 + 16 + (hi ? 8 : 0));
        }
    }

    float run_max[8], run_sum[8];
    v8f o[4];
#pragma unroll
    for (int i = 0; i < 8; i++) { run_max[i] = -3.0e38f; run_sum[i] = 0.0f; }
#pragma unroll
    for (int j = 0; j < 4; j++)
#pragma unroll
        for (int i = 0; i < 8; i++) o[j][i] = 0.0f;

    unsigned short* pw = &lds_p[wave * 512];

    for (int n0 = 0; n0 < Nn; n0 += 32) {
        // ---- K fragments for this chunk (B element = K[n][d], contig in d) -
        BF16Frag kf[2][2];
#pragma unroll
        for (int sn = 0; sn < 2; sn++) {
            const unsigned short* krow =
                kb + ((size_t)bh * Nn + n0 + sn * 16 + ln) * Dd;
#pragma unroll
            for (int f = 0; f < 2; f++) {
                kf[sn][f].h[0] = *(const v8bf*)(krow + f * 32 + (hi ? 8 : 0));
                kf[sn][f].h[1] = *(const v8bf*)(krow + f * 32 + 16 + (hi ? 8 : 0));
            }
        }
        // ---- S = (Q*scale) . K^T ----
        v8f s[2];
#pragma unroll
        for (int sn = 0; sn < 2; sn++) {
#pragma unroll
            for (int i = 0; i < 8; i++) s[sn][i] = 0.0f;
#pragma unroll
            for (int f = 0; f < 2; f++)
                s[sn] = __builtin_amdgcn_wmma_f32_16x16x32_bf16(
                    false, aq[f].v, false, kf[sn][f].v, (short)0, s[sn], false,
                    false);
        }

        // ---- V fragments (pre-transposed [b,h,d,n]: contig run over n) ----
        // issued here so the loads hide under the softmax VALU/TRANS chain
        BF16Frag vf[4];
#pragma unroll
        for (int j = 0; j < 4; j++) {
            const unsigned short* vrow =
                vb + ((size_t)bh * Dd + j * 16 + ln) * Nn + n0 + (hi ? 8 : 0);
            vf[j].h[0] = *(const v8bf*)(vrow);
            vf[j].h[1] = *(const v8bf*)(vrow + 16);
        }

        // ---- online softmax (rows live across one 16-lane half; DPP) ----
#pragma unroll
        for (int i = 0; i < 8; i++) {
            float t = dpp_max16(fmaxf(s[0][i], s[1][i]));
            float nm   = fmaxf(run_max[i], t);
            float corr = __expf(run_max[i] - nm);
            run_max[i] = nm;
            float p0 = __expf(s[0][i] - nm);
            float p1 = __expf(s[1][i] - nm);
            float rs = dpp_add16(p0 + p1);
            run_sum[i] = run_sum[i] * corr + rs;
#pragma unroll
            for (int j = 0; j < 4; j++) o[j][i] *= corr;
            s[0][i] = p0;
            s[1][i] = p1;
        }

        // ---- C-layout -> A-layout for P via per-wave LDS strip ----
#pragma unroll
        for (int i = 0; i < 8; i++) {
            int row = hi ? i + 8 : i;
            pw[row * 32 + ln]      = f2bf(s[0][i]);
            pw[row * 32 + 16 + ln] = f2bf(s[1][i]);
        }
        // wave-local visibility: DS ops complete in order; wave32 lockstep
        asm volatile("s_wait_dscnt 0x0" ::: "memory");

        BF16Frag pa;
        pa.h[0] = *(const v8bf*)&pw[ln * 32 + (hi ? 8 : 0)];
        pa.h[1] = *(const v8bf*)&pw[ln * 32 + 16 + (hi ? 8 : 0)];
        asm volatile("s_wait_dscnt 0x0" ::: "memory");

        // ---- O += P @ V ----
#pragma unroll
        for (int j = 0; j < 4; j++)
            o[j] = __builtin_amdgcn_wmma_f32_16x16x32_bf16(
                false, pa.v, false, vf[j].v, (short)0, o[j], false, false);
    }

    // ---- finalize: O / rowsum, store bf16 to [b][m][h*64+d] ----
    int b = bh >> 4, h = bh & 15;
    float inv[8];
#pragma unroll
    for (int i = 0; i < 8; i++) inv[i] = __builtin_amdgcn_rcpf(run_sum[i]);
#pragma unroll
    for (int j = 0; j < 4; j++) {
#pragma unroll
        for (int i = 0; i < 8; i++) {
            int row = m0 + wave * 16 + (hi ? i + 8 : i);
            float val = o[j][i] * inv[i];
            int col = h * Dd + j * 16 + ln;
            ab[(size_t)(b * Mm + row) * Ee + col] = f2bf(val);
        }
    }
}

// ---------------------------------------------------------------------------
extern "C" void kernel_launch(void* const* d_in, const int* in_sizes, int n_in,
                              void* d_out, int out_size, void* d_ws,
                              size_t ws_size, hipStream_t stream) {
    const float* query = (const float*)d_in[0];
    const float* key   = (const float*)d_in[1];
    const float* value = (const float*)d_in[2];
    const float* Wq = (const float*)d_in[3];
    const float* bq = (const float*)d_in[4];
    const float* Wk = (const float*)d_in[5];
    const float* bk = (const float*)d_in[6];
    const float* Wv = (const float*)d_in[7];
    const float* bv = (const float*)d_in[8];
    const float* Wo = (const float*)d_in[9];
    const float* bo = (const float*)d_in[10];

    // bf16 workspace: Q,K in [b,h,seq,d]; V in [b,h,d,seq]; attn-out [b,m,E]
    const size_t seg = (size_t)Bc * Hh * Mm * Dd;  // 8,388,608 elems (16 MB)
    unsigned short* qb  = (unsigned short*)d_ws;
    unsigned short* kb_ = qb + seg;
    unsigned short* vb_ = kb_ + seg;
    unsigned short* ab_ = vb_ + seg;

    dim3 gridL(Ee / 64, (Bc * Mm) / 128);  // (16, 64)
    dim3 blk(256);

    linear_wmma<0><<<gridL, blk, 0, stream>>>(query, Wq, bq, qb, SCALE);
    linear_wmma<0><<<gridL, blk, 0, stream>>>(key, Wk, bk, kb_, 1.0f);
    linear_wmma<2><<<gridL, blk, 0, stream>>>(value, Wv, bv, vb_, 1.0f);

    flash_attn<<<dim3(Bc * Hh * (Mm / 128)), blk, 0, stream>>>(qb, kb_, vb_, ab_);

    linear_wmma<1><<<gridL, blk, 0, stream>>>(ab_, Wo, bo, d_out, 1.0f);
}
